// GGRU_ABIDE_32908039422441
// MI455X (gfx1250) — matrix operand order repaired and checked by reading, not compiled, not run on hardware
//
#include <hip/hip_runtime.h>
#include <cstdint>

#define N_NODES 113664
#define FEAT 128
#define N_EDGES 909312
#define T_STEPS 4
#define N_GRAPHS 1024
#define NODES_PER_GRAPH 111

// Set to 0 if the assembler rejects the async-to-LDS mnemonics.
#define USE_ASYNC_LDS 1

typedef __attribute__((ext_vector_type(16))) __bf16    v16bf;
typedef __attribute__((ext_vector_type(8)))  float     v8f;
typedef __attribute__((ext_vector_type(4)))  unsigned  u32x4;   // 16B aligned

struct Frag2 { u32x4 lo, hi; };

static __device__ __forceinline__ unsigned short f2bf(float f) {
  unsigned u = __builtin_bit_cast(unsigned, f);
  u += 0x7FFFu + ((u >> 16) & 1u);          // round-to-nearest-even
  return (unsigned short)(u >> 16);
}
static __device__ __forceinline__ unsigned pack2bf(float a, float b) {
  return (unsigned)f2bf(a) | ((unsigned)f2bf(b) << 16);
}
static __device__ __forceinline__ v16bf fragFrom(u32x4 lo, u32x4 hi) {
  Frag2 q{lo, hi};
  return __builtin_bit_cast(v16bf, q);
}
static __device__ __forceinline__ float sigmoidf_fast(float x) {
  return 1.0f / (1.0f + __expf(-x));
}

// Async DMA: copy `dwords` dwords from global `src` into LDS `dstLds`,
// 16B per lane per issue, tracked on ASYNCcnt (global_load_async_to_lds_b128).
static __device__ __forceinline__ void asyncCopyToLds(unsigned* dstLds,
                                                      const unsigned* src,
                                                      int dwords, int tid) {
#if USE_ASYNC_LDS
  for (int i = tid * 4; i < dwords; i += 256 * 4) {
    unsigned lds = (unsigned)(uintptr_t)(dstLds + i);   // flat->local: low 32 bits
    unsigned off = (unsigned)i * 4u;                    // byte offset from SGPR base
    asm volatile("global_load_async_to_lds_b128 %0, %1, %2"
                 :: "v"(lds), "v"(off), "s"(src) : "memory");
  }
#else
  for (int i = tid; i < dwords / 4; i += 256)
    ((u32x4*)dstLds)[i] = ((const u32x4*)src)[i];
#endif
}
static __device__ __forceinline__ void asyncWait() {
#if USE_ASYNC_LDS
  asm volatile("s_wait_asynccnt 0" ::: "memory");
#endif
}

// ---------------------------------------------------------------------------
// Shared WMMA core: one wave computes a 32x64 tile of  A[MBx256] @ W[256xNC].
// aLds: [row][kpair] packed bf16 pairs, row stride 128 dwords (K = 256).
// wLds: [col][kpair] packed bf16 pairs, col stride 128 dwords.
// Layouts per CDNA5 ISA 7.12.2 (16-bit A 16x32, B 32x16).
// ---------------------------------------------------------------------------
static __device__ __forceinline__ void wmmaCore(
    const unsigned* __restrict__ wLds, const unsigned* __restrict__ aLds,
    int mb, int nb, int lane, v8f acc[2][4]) {
  const int half = lane >> 4;
  const int l15  = lane & 15;
#pragma unroll
  for (int kb = 0; kb < 8; ++kb) {
    const int kpb = kb * 16;                       // 16 kpairs = K step of 32
    v16bf aF[2];
#pragma unroll
    for (int i = 0; i < 2; ++i) {
      const u32x4* p = (const u32x4*)(aLds + (mb + i * 16 + l15) * 128 + kpb + half * 4);
      aF[i] = fragFrom(p[0], p[2]);                // kpairs {0..3},{8..11} (+half*4)
    }
    v16bf bF[4];
#pragma unroll
    for (int j = 0; j < 4; ++j) {
      const u32x4* p = (const u32x4*)(wLds + (nb + j * 16 + l15) * 128 + kpb + half * 4);
      bF[j] = fragFrom(p[0], p[2]);                // kpairs {0..3},{8..11} (+half*4)
    }
#pragma unroll
    for (int i = 0; i < 2; ++i)
#pragma unroll
      for (int j = 0; j < 4; ++j)
        acc[i][j] = __builtin_amdgcn_wmma_f32_16x16x32_bf16(
            false, aF[i], false, bF[j], (short)0, acc[i][j], false, false);
  }
}

// ---------------------------------------------------------------------------
// Prep: pack weights into bf16-pair, B-fragment-friendly layout; fuse biases.
// ---------------------------------------------------------------------------
__global__ void packPrepKernel(
    const float* __restrict__ Wzi, const float* __restrict__ bzi,
    const float* __restrict__ Wzh, const float* __restrict__ bzh,
    const float* __restrict__ Wri, const float* __restrict__ bri,
    const float* __restrict__ Wrh, const float* __restrict__ brh,
    const float* __restrict__ Whi, const float* __restrict__ bhi,
    const float* __restrict__ Whh, const float* __restrict__ bhh,
    unsigned* __restrict__ wpackZR, unsigned* __restrict__ wpackH,
    float* __restrict__ biasZR, float* __restrict__ biasH) {
  int idx = blockIdx.x * 256 + threadIdx.x;
  if (idx < 32768) {                                  // wpackZR[n][kp], n in [0,256)
    int n = idx >> 7, kp = idx & 127;
    const float *Wa, *Wb; int col;
    if (n < 128) { Wa = Wzi; Wb = Wzh; col = n; }
    else         { Wa = Wri; Wb = Wrh; col = n - 128; }
    const float* W = (kp < 64) ? Wa : Wb;
    int r0 = (kp < 64) ? kp * 2 : kp * 2 - 128;
    wpackZR[idx] = pack2bf(W[r0 * 128 + col], W[(r0 + 1) * 128 + col]);
  } else if (idx < 49152) {                           // wpackH[n][kp], n in [0,128)
    int j = idx - 32768; int n = j >> 7, kp = j & 127;
    const float* W = (kp < 64) ? Whi : Whh;
    int r0 = (kp < 64) ? kp * 2 : kp * 2 - 128;
    wpackH[j] = pack2bf(W[r0 * 128 + n], W[(r0 + 1) * 128 + n]);
  } else if (idx < 49408) {
    int n = idx - 49152;
    biasZR[n] = (n < 128) ? (bzi[n] + bzh[n]) : (bri[n - 128] + brh[n - 128]);
  } else if (idx < 49536) {
    int n = idx - 49408;
    biasH[n] = bhi[n] + bhh[n];
  }
}

// h0 = eye(N, 128)
__global__ void hInitKernel(float* __restrict__ h) {
  int idx = blockIdx.x * 256 + threadIdx.x;           // over N*32 float4s
  int row = idx >> 5, q = idx & 31;
  int c0 = q * 4;
  float4 v;
  v.x = (row == c0 + 0) ? 1.f : 0.f;
  v.y = (row == c0 + 1) ? 1.f : 0.f;
  v.z = (row == c0 + 2) ? 1.f : 0.f;
  v.w = (row == c0 + 3) ? 1.f : 0.f;
  ((float4*)h)[idx] = v;
}

__global__ void degInitKernel(float* __restrict__ deg) {
  int i = blockIdx.x * 256 + threadIdx.x;
  deg[i] = 1.0f;                                      // self-loop weight
}

__global__ void degAccumKernel(float* __restrict__ deg,
                               const int* __restrict__ dst,
                               const float* __restrict__ ew) {
  int e = blockIdx.x * 256 + threadIdx.x;
  unsafeAtomicAdd(&deg[dst[e]], ew[e]);
}

// px = x / deg  (self-loop term of the propagation)
__global__ void pxInitKernel(float* __restrict__ px, const float* __restrict__ x,
                             const float* __restrict__ deg) {
  int idx = blockIdx.x * 256 + threadIdx.x;           // over N*32 float4s
  int row = idx >> 5;
  float inv = 1.0f / deg[row];
  float4 xv = ((const float4*)x)[idx];
  float4 o; o.x = xv.x * inv; o.y = xv.y * inv; o.z = xv.z * inv; o.w = xv.w * inv;
  ((float4*)px)[idx] = o;
}

// px[dst] += rsqrt(deg[src]) * ew * rsqrt(deg[dst]) * x[src]; one wave per edge
__global__ void edgeScatterKernel(float* __restrict__ px, const float* __restrict__ x,
                                  const float* __restrict__ deg,
                                  const int* __restrict__ src, const int* __restrict__ dst,
                                  const float* __restrict__ ew) {
  int e = blockIdx.x * 8 + (threadIdx.x >> 5);
  int lane = threadIdx.x & 31;
  int s = src[e], d = dst[e];
  float norm = __frsqrt_rn(deg[s]) * ew[e] * __frsqrt_rn(deg[d]);
  const float4 xv = ((const float4*)(x + (size_t)s * FEAT))[lane];
  float* pd = px + (size_t)d * FEAT + lane * 4;
  unsafeAtomicAdd(pd + 0, norm * xv.x);
  unsafeAtomicAdd(pd + 1, norm * xv.y);
  unsafeAtomicAdd(pd + 2, norm * xv.z);
  unsafeAtomicAdd(pd + 3, norm * xv.w);
}

// ---------------------------------------------------------------------------
// G1: [px|h] @ [[Wzi,Wri],[Wzh,Wrh]] -> z = sigmoid(.), hr = r * h
// Block: 64 rows x 256 cols, 8 waves (2 along M x 4 along N).
// ---------------------------------------------------------------------------
__global__ __launch_bounds__(256) void gruGateKernel(
    const float* __restrict__ px, const float* __restrict__ h,
    const unsigned* __restrict__ wpackZR, const float* __restrict__ biasZR,
    float* __restrict__ zOut, float* __restrict__ hrOut) {
  extern __shared__ unsigned smem[];
  unsigned* wLds = smem;                      // 256*128 dwords = 128 KB
  unsigned* aLds = wLds + 256 * 128;          // 64*128  dwords = 32 KB
  float*    bLds = (float*)(aLds + 64 * 128); // 256 floats

  const int tid = threadIdx.x;
  const int rowBase = blockIdx.x * 64;

  asyncCopyToLds(wLds, wpackZR, 256 * 128, tid);       // async DMA weights -> LDS
  if (tid < 256) bLds[tid] = biasZR[tid];
  for (int i = tid; i < 64 * 128; i += 256) {          // overlap: pack A while DMA runs
    int r = i >> 7, kp = i & 127;
    const float* rowp = (kp < 64) ? (px + (size_t)(rowBase + r) * FEAT)
                                  : (h  + (size_t)(rowBase + r) * FEAT);
    float2 v = ((const float2*)rowp)[kp & 63];
    aLds[i] = pack2bf(v.x, v.y);
  }
  asyncWait();
  __syncthreads();

  const int w = tid >> 5, lane = tid & 31;
  const int mb = (w >> 2) * 32, nb = (w & 3) * 64;
  v8f acc[2][4];
  const v8f vz = {0.f, 0.f, 0.f, 0.f, 0.f, 0.f, 0.f, 0.f};
#pragma unroll
  for (int i = 0; i < 2; ++i)
#pragma unroll
    for (int j = 0; j < 4; ++j) acc[i][j] = vz;

  wmmaCore(wLds, aLds, mb, nb, lane, acc);

  const int half = lane >> 4, l15 = lane & 15;
#pragma unroll
  for (int i = 0; i < 2; ++i)
#pragma unroll
    for (int j = 0; j < 4; ++j)
#pragma unroll
      for (int v = 0; v < 8; ++v) {
        int row = rowBase + mb + i * 16 + v + half * 8;
        int col = nb + j * 16 + l15;
        float s = sigmoidf_fast(acc[i][j][v] + bLds[col]);
        if (col < FEAT) {
          zOut[(size_t)row * FEAT + col] = s;
        } else {
          int c = col - FEAT;
          hrOut[(size_t)row * FEAT + c] = s * h[(size_t)row * FEAT + c];
        }
      }
}

// ---------------------------------------------------------------------------
// G2: cand = tanh([px|hr] @ [[Whi],[Whh]] + b); h = (1-z)*hr;
//     sum += h + z*cand.  Block: 128 rows x 128 cols, 8 waves (4 x 2).
// ---------------------------------------------------------------------------
__global__ __launch_bounds__(256) void gruCandKernel(
    const float* __restrict__ px, const float* __restrict__ hr,
    const unsigned* __restrict__ wpackH, const float* __restrict__ biasH,
    const float* __restrict__ zBuf, float* __restrict__ hBuf,
    float* __restrict__ sumOut, int first) {
  extern __shared__ unsigned smem[];
  unsigned* wLds = smem;                        // 128*128 dwords = 64 KB
  unsigned* aLds = wLds + 128 * 128;            // 128*128 dwords = 64 KB
  float*    bLds = (float*)(aLds + 128 * 128);  // 128 floats

  const int tid = threadIdx.x;
  const int rowBase = blockIdx.x * 128;

  asyncCopyToLds(wLds, wpackH, 128 * 128, tid);        // async DMA weights -> LDS
  if (tid < 128) bLds[tid] = biasH[tid];
  for (int i = tid; i < 128 * 128; i += 256) {
    int r = i >> 7, kp = i & 127;
    const float* rowp = (kp < 64) ? (px + (size_t)(rowBase + r) * FEAT)
                                  : (hr + (size_t)(rowBase + r) * FEAT);
    float2 v = ((const float2*)rowp)[kp & 63];
    aLds[i] = pack2bf(v.x, v.y);
  }
  asyncWait();
  __syncthreads();

  const int w = tid >> 5, lane = tid & 31;
  const int mb = (w >> 1) * 32, nb = (w & 1) * 64;
  v8f acc[2][4];
  const v8f vz = {0.f, 0.f, 0.f, 0.f, 0.f, 0.f, 0.f, 0.f};
#pragma unroll
  for (int i = 0; i < 2; ++i)
#pragma unroll
    for (int j = 0; j < 4; ++j) acc[i][j] = vz;

  wmmaCore(wLds, aLds, mb, nb, lane, acc);

  const int half = lane >> 4, l15 = lane & 15;
#pragma unroll
  for (int i = 0; i < 2; ++i)
#pragma unroll
    for (int j = 0; j < 4; ++j)
#pragma unroll
      for (int v = 0; v < 8; ++v) {
        int row = rowBase + mb + i * 16 + v + half * 8;
        int col = nb + j * 16 + l15;
        size_t idx = (size_t)row * FEAT + col;
        float cand = tanhf(acc[i][j][v] + bLds[col]);
        float zv = zBuf[idx];
        float hrv = hr[idx];
        float hnew = (1.0f - zv) * hrv;
        float out = hnew + zv * cand;
        hBuf[idx] = hnew;
        sumOut[idx] = first ? out : (sumOut[idx] + out);
      }
}

// mean-pool (111 contiguous nodes per graph) + 128->2 linear head
__global__ void poolLinKernel(const float* __restrict__ sumOut,
                              const float* __restrict__ Wlin,
                              const float* __restrict__ blin,
                              float* __restrict__ out) {
  __shared__ float r0[128], r1[128];
  int g = blockIdx.x, f = threadIdx.x;
  const float* base = sumOut + (size_t)g * NODES_PER_GRAPH * FEAT + f;
  float s = 0.f;
  for (int i = 0; i < NODES_PER_GRAPH; ++i) s += base[(size_t)i * FEAT];
  float p = s * (1.0f / (float)NODES_PER_GRAPH);
  r0[f] = p * Wlin[f * 2 + 0];
  r1[f] = p * Wlin[f * 2 + 1];
  __syncthreads();
  for (int st = 64; st > 0; st >>= 1) {
    if (f < st) { r0[f] += r0[f + st]; r1[f] += r1[f + st]; }
    __syncthreads();
  }
  if (f == 0) {
    out[g * 2 + 0] = r0[0] + blin[0];
    out[g * 2 + 1] = r1[0] + blin[1];
  }
}

extern "C" void kernel_launch(void* const* d_in, const int* in_sizes, int n_in,
                              void* d_out, int out_size, void* d_ws, size_t ws_size,
                              hipStream_t stream) {
  const float* xs  = (const float*)d_in[0];   // [T,N,F]
  const float* eas = (const float*)d_in[1];   // [T,E]
  const int*   eis = (const int*)d_in[2];     // [T,2,E]
  // d_in[3] = batch (contiguous runs of 111, unused)
  const float* Wzi = (const float*)d_in[4];  const float* bzi = (const float*)d_in[5];
  const float* Wzh = (const float*)d_in[6];  const float* bzh = (const float*)d_in[7];
  const float* Wri = (const float*)d_in[8];  const float* bri = (const float*)d_in[9];
  const float* Wrh = (const float*)d_in[10]; const float* brh = (const float*)d_in[11];
  const float* Whi = (const float*)d_in[12]; const float* bhi = (const float*)d_in[13];
  const float* Whh = (const float*)d_in[14]; const float* bhh = (const float*)d_in[15];
  const float* Wlin = (const float*)d_in[16]; const float* blin = (const float*)d_in[17];
  float* out = (float*)d_out;

  const size_t NF = (size_t)N_NODES * FEAT;
  float* ws   = (float*)d_ws;
  float* h    = ws;
  float* px   = ws + 1 * NF;
  float* z    = ws + 2 * NF;
  float* hr   = ws + 3 * NF;
  float* sum  = ws + 4 * NF;
  float* deg  = ws + 5 * NF;                          // N floats
  unsigned* wpackZR = (unsigned*)(deg + N_NODES);     // 256*128
  unsigned* wpackH  = wpackZR + 256 * 128;            // 128*128
  float* biasZR = (float*)(wpackH + 128 * 128);       // 256
  float* biasH  = biasZR + 256;                       // 128

  packPrepKernel<<<194, 256, 0, stream>>>(Wzi, bzi, Wzh, bzh, Wri, bri, Wrh, brh,
                                          Whi, bhi, Whh, bhh,
                                          wpackZR, wpackH, biasZR, biasH);
  hInitKernel<<<N_NODES * 32 / 256, 256, 0, stream>>>(h);

  const size_t lds1 = (size_t)(256 * 128 + 64 * 128) * 4 + 256 * 4;   // ~161 KB
  const size_t lds2 = (size_t)(128 * 128 + 128 * 128) * 4 + 128 * 4;  // ~128.5 KB

  for (int t = 0; t < T_STEPS; ++t) {
    const float* x  = xs + (size_t)t * NF;
    const float* ew = eas + (size_t)t * N_EDGES;
    const int* src  = eis + (size_t)t * 2 * N_EDGES;
    const int* dst  = src + N_EDGES;

    degInitKernel  <<<N_NODES / 256, 256, 0, stream>>>(deg);
    degAccumKernel <<<N_EDGES / 256, 256, 0, stream>>>(deg, dst, ew);
    pxInitKernel   <<<N_NODES * 32 / 256, 256, 0, stream>>>(px, x, deg);
    edgeScatterKernel<<<N_EDGES / 8, 256, 0, stream>>>(px, x, deg, src, dst, ew);
    gruGateKernel  <<<N_NODES / 64, 256, lds1, stream>>>(px, h, wpackZR, biasZR, z, hr);
    gruCandKernel  <<<N_NODES / 128, 256, lds2, stream>>>(px, hr, wpackH, biasH, z, h,
                                                          sum, (t == 0) ? 1 : 0);
  }

  poolLinKernel<<<N_GRAPHS, 128, 0, stream>>>(sum, Wlin, blin, out);
}
